// MultiHeadAttention_36318243455587
// MI455X (gfx1250) — compile-verified, matrix operand order checked by reading, not run on hardware
//
#include <hip/hip_runtime.h>
#include <cstdint>
#include <cstddef>

// ---------------------------------------------------------------------------
// Problem constants (match reference)
// ---------------------------------------------------------------------------
#define DIM       4096
#define N_HEADS   32
#define N_KV      8
#define HEAD_DIM  128
#define BATCH     2
#define SEQLEN    2048
#define TOKENS    (BATCH * SEQLEN)        // 4096
#define KV_DIM    (N_KV * HEAD_DIM)       // 1024

// ---------------------------------------------------------------------------
// CDNA5 WMMA types / helpers (wave32, 16x16x32 bf16 -> f32)
// ---------------------------------------------------------------------------
typedef __attribute__((ext_vector_type(16))) __bf16       v16bf;
typedef __attribute__((ext_vector_type(8)))  float        v8f;
typedef __attribute__((ext_vector_type(4)))  unsigned int u32x4;
typedef __attribute__((ext_vector_type(8)))  unsigned int u32x8;

static __device__ __forceinline__ unsigned short f2bf(float f) {
  unsigned int u = __float_as_uint(f);
  unsigned int r = 0x7FFFu + ((u >> 16) & 1u);   // round-to-nearest-even
  return (unsigned short)((u + r) >> 16);
}

static __device__ __forceinline__ v8f zero8() {
  v8f z = {0.f, 0.f, 0.f, 0.f, 0.f, 0.f, 0.f, 0.f};
  return z;
}

// Load one 16x32 bf16 fragment from a row-major matrix with leading dim `ld`
// (elements). CDNA5 A-operand layout: lane L holds row (L&15); K elements
// split as [grp*8 .. grp*8+7] and [16+grp*8 .. 16+grp*8+7] with grp = L>>4.
// The B-operand layout mirrors this with N in place of M, so the same loader
// serves B when the matrix is stored N-major (row = output column).
static __device__ __forceinline__ v16bf load_frag(const unsigned short* p, int ld) {
  const int lane = threadIdx.x & 31;
  const int row  = lane & 15;
  const int grp  = lane >> 4;
  const unsigned short* q = p + (size_t)row * (size_t)ld + grp * 8;
  union { v16bf bf; u32x4 u[2]; } f;
  f.u[0] = *(const u32x4*)(q);
  f.u[1] = *(const u32x4*)(q + 16);
  return f.bf;
}

static __device__ __forceinline__ v8f wmma_bf16(v16bf a, v16bf b, v8f c) {
  // (neg_a, A, neg_b, B, c_mod, C, reuse_a, reuse_b)
  return __builtin_amdgcn_wmma_f32_16x16x32_bf16(false, a, false, b, (short)0, c,
                                                 false, false);
}

// ---------------------------------------------------------------------------
// Tensor Data Mover: DMA a 2D bf16 tile (tile_d0 x tile_d1 elements) from a
// row-major global tensor (leading dim = stride0 elements) into LDS.
// D# layout per CDNA5 ISA sect. 8.3/8.4:
//   group0: count=1 | lds_addr | global_addr[56:0] | type=2
//   group1: data_size=2B | tensor_dim0/1 | tile_dim0/1 | tensor_dim0_stride
//   groups 2/3: zero (2D tensor; dims 2+ unused)
// Issued via inline asm (portable across ROCm7.2 / amdgpu-toolchain builtin
// arity split).  Tracked with TENSORcnt.
// ---------------------------------------------------------------------------
static __device__ __forceinline__ void tdm_load_2d_bf16(
    unsigned lds_off, const void* gptr,
    unsigned tensor_d0, unsigned tensor_d1,
    unsigned tile_d0, unsigned tile_d1, unsigned stride0) {
  unsigned long long ga = (unsigned long long)gptr;
  unsigned ga_lo = __builtin_amdgcn_readfirstlane((unsigned)ga);
  unsigned ga_hi = __builtin_amdgcn_readfirstlane((unsigned)(ga >> 32));
  unsigned lds   = __builtin_amdgcn_readfirstlane(lds_off);

  u32x4 g0;
  g0[0] = 1u;                                       // count=1 user descriptor
  g0[1] = lds;                                      // lds_addr (bytes)
  g0[2] = ga_lo;                                    // global_addr[31:0]
  g0[3] = (ga_hi & 0x01FFFFFFu) | (2u << 30);       // addr[56:32] | type=2

  u32x8 g1;
  g1[0] = (1u << 16);                               // data_size=1 -> 2 bytes
  g1[1] = (tensor_d0 & 0xFFFFu) << 16;              // tensor_dim0 @ bit48
  g1[2] = (tensor_d0 >> 16) | ((tensor_d1 & 0xFFFFu) << 16);   // @bit64/80
  g1[3] = (tensor_d1 >> 16) | ((tile_d0 & 0xFFFFu) << 16);     // tile_dim0
  g1[4] = tile_d1 & 0xFFFFu;                        // tile_dim1 (tile_dim2=0)
  g1[5] = stride0;                                  // dim0 stride [31:0]
  g1[6] = 0u;                                       // stride0 hi | stride1 lo
  g1[7] = 0u;                                       // stride1 hi

  u32x4 gz = {0u, 0u, 0u, 0u};                      // groups 2/3 (2D: unused)

  asm volatile("tensor_load_to_lds %0, %1, %2, %3"
               :
               : "s"(g0), "s"(g1), "s"(gz), "s"(gz)
               : "memory");
}

// ---------------------------------------------------------------------------
// Kernel 1: fp32 -> bf16 conversion (grid-stride)
// ---------------------------------------------------------------------------
__global__ void cvt_f32_bf16(const float* __restrict__ in,
                             unsigned short* __restrict__ out, long long n) {
  long long i = (long long)blockIdx.x * blockDim.x + threadIdx.x;
  long long stride = (long long)gridDim.x * blockDim.x;
  for (; i < n; i += stride) out[i] = f2bf(in[i]);
}

// ---------------------------------------------------------------------------
// Kernel 2: bf16 GEMM  C[M,N] = A[M,K] * B[N,K]^T  (fp32 out)
// One wave computes a 64x64 C tile: 4 M-frags x 4 N-frags = 16 accumulators
// (128 VGPRs) + 8 operand frags.  __launch_bounds__(128,1) -> no spills.
// A TDM tensor_load_to_lds stages the first B panel into LDS as a hardware
// prefetch before the K loop (TENSORcnt / s_wait_tensorcnt path).
// ---------------------------------------------------------------------------
__global__ __launch_bounds__(128, 1) void gemm_bf16(
    const unsigned short* __restrict__ A,
    const unsigned short* __restrict__ B,
    float* __restrict__ C, int M, int N, int K) {
  const int wave = blockIdx.x * (blockDim.x >> 5) + (threadIdx.x >> 5);
  const int ntiles = N >> 6;                      // 64-wide tiles
  const int mt = wave / ntiles;
  const int nt = wave % ntiles;
  if (mt * 64 >= M) return;
  const int lane = threadIdx.x & 31;

  __shared__ __align__(16) unsigned short stage[4][64 * 32];  // 16 KB/block
  unsigned short* sb = stage[(threadIdx.x >> 5) & 3];

  v8f acc[4][4];
  for (int i = 0; i < 4; ++i)
    for (int j = 0; j < 4; ++j) acc[i][j] = zero8();

  const unsigned short* Abase = A + (size_t)(mt * 64) * K;
  const unsigned short* Bbase = B + (size_t)(nt * 64) * K;

  // TDM: DMA the first 64x32 B panel (64 rows of 32 bf16) into this wave's
  // LDS stage buffer; acts as a panel prefetch ahead of the vector loads.
  tdm_load_2d_bf16((unsigned)(unsigned long long)sb, Bbase,
                   (unsigned)K, 64u, 32u, 64u, (unsigned)K);
  __builtin_amdgcn_s_wait_tensorcnt(0);

  for (int k0 = 0; k0 < K; k0 += 32) {
    if (k0 + 32 < K) {                            // -> global_prefetch_b8
      __builtin_prefetch(Abase + k0 + 32, 0, 3);
      __builtin_prefetch(Bbase + k0 + 32, 0, 3);
    }
    v16bf a[4], b[4];
    for (int i = 0; i < 4; ++i)
      a[i] = load_frag(Abase + (size_t)(i * 16) * K + k0, K);
    for (int j = 0; j < 4; ++j)
      b[j] = load_frag(Bbase + (size_t)(j * 16) * K + k0, K);
    for (int i = 0; i < 4; ++i)
      for (int j = 0; j < 4; ++j)
        acc[i][j] = wmma_bf16(a[i], b[j], acc[i][j]);
  }

  // C/D layout: vgpr r -> row r (lanes 0-15) or r+8 (lanes 16-31); col = lane&15
  const int grp = lane >> 4, ln = lane & 15;
  for (int i = 0; i < 4; ++i)
    for (int j = 0; j < 4; ++j)
      for (int r = 0; r < 8; ++r) {
        int m = mt * 64 + i * 16 + r + 8 * grp;
        int n = nt * 64 + j * 16 + ln;
        C[(size_t)m * N + n] = acc[i][j][r];
      }
}

// ---------------------------------------------------------------------------
// Kernel 3: RoPE (fp32 in) + bf16 out.  Layout [tok][head][head_dim].
// One thread per (even,odd) pair.
// ---------------------------------------------------------------------------
__global__ void rope_cvt(const float* __restrict__ in,
                         const float* __restrict__ freqs,   // [S][64][2]
                         unsigned short* __restrict__ out,
                         int heads, long long npairs) {
  long long idx = (long long)blockIdx.x * blockDim.x + threadIdx.x;
  if (idx >= npairs) return;
  const int hd2 = HEAD_DIM / 2;                   // 64
  long long ppt = (long long)heads * hd2;         // pairs per token
  long long tok = idx / ppt;
  int rem = (int)(idx % ppt);
  int i = rem & (hd2 - 1);                        // pair index within head
  int s = (int)(tok % SEQLEN);
  float fr = freqs[((size_t)s * hd2 + i) * 2 + 0];
  float fi = freqs[((size_t)s * hd2 + i) * 2 + 1];
  float xr = in[idx * 2 + 0];
  float xi = in[idx * 2 + 1];
  out[idx * 2 + 0] = f2bf(xr * fr - xi * fi);
  out[idx * 2 + 1] = f2bf(xi * fr + xr * fi);
}

// ---------------------------------------------------------------------------
// Kernel 4: V fp32 [b][s][kvh][d]  ->  bf16 transposed [b][kvh][d][s]
// so PV's B-operand fragments are contiguous (ld = SEQLEN).
// ---------------------------------------------------------------------------
__global__ void v_transpose_cvt(const float* __restrict__ in,
                                unsigned short* __restrict__ out) {
  long long idx = (long long)blockIdx.x * blockDim.x + threadIdx.x;
  long long total = (long long)TOKENS * KV_DIM;
  if (idx >= total) return;
  long long tok = idx >> 10;                      // /KV_DIM
  int e = (int)(idx & (KV_DIM - 1));
  int b = (int)(tok / SEQLEN), s = (int)(tok % SEQLEN);
  int kvh = e >> 7, d = e & 127;
  out[((size_t)(b * N_KV + kvh) * HEAD_DIM + d) * SEQLEN + s] = f2bf(in[idx]);
}

// ---------------------------------------------------------------------------
// Kernel 5: causal flash attention, one wave per (batch, head, 16-row Q tile).
// Q bf16 [tok][4096], K bf16 [tok][1024], Vt bf16 [b][kvh][128][S].
// ---------------------------------------------------------------------------
__global__ __launch_bounds__(128, 2) void flash_attn(
    const unsigned short* __restrict__ Qb,
    const unsigned short* __restrict__ Kb,
    const unsigned short* __restrict__ Vt,
    unsigned short* __restrict__ Ob) {
  const int wave = blockIdx.x * (blockDim.x >> 5) + (threadIdx.x >> 5);
  const int qtiles = SEQLEN / 16;                 // 128
  const int qt = wave % qtiles;
  const int h  = (wave / qtiles) % N_HEADS;
  const int b  = wave / (qtiles * N_HEADS);
  if (b >= BATCH) return;
  const int lane = threadIdx.x & 31;
  const int grp = lane >> 4, ln = lane & 15;
  const int kvh = h >> 2;                         // GQA: 4 Q heads per KV head
  const int q0 = qt * 16;
  const float scale = 0.08838834764831845f;       // 1/sqrt(128)

  __shared__ __align__(16) unsigned short p_lds[4][16 * 32]; // per-wave P tile
  unsigned short* pbuf = p_lds[(threadIdx.x >> 5) & 3];

  // Q block 16x128 -> 4 A fragments, resident for the whole key loop
  const unsigned short* Qbase = Qb + (size_t)(b * SEQLEN + q0) * DIM + h * HEAD_DIM;
  v16bf qf[4];
  for (int f = 0; f < 4; ++f) qf[f] = load_frag(Qbase + f * 32, DIM);

  v8f o[8];
  for (int j = 0; j < 8; ++j) o[j] = zero8();
  float mrun[8], lrun[8];
  for (int r = 0; r < 8; ++r) { mrun[r] = -1e30f; lrun[r] = 0.f; }

  const int kend = q0 + 16;                       // causal horizon
  for (int k0 = 0; k0 < kend; k0 += 32) {
    // ---- S = Q * K^T for 32 keys (two 16-key column blocks) -------------
    v8f sacc[2] = {zero8(), zero8()};
    for (int nb = 0; nb < 2; ++nb) {
      const unsigned short* Kbase =
          Kb + (size_t)(b * SEQLEN + k0 + nb * 16) * KV_DIM + kvh * HEAD_DIM;
      for (int f = 0; f < 4; ++f)
        sacc[nb] = wmma_bf16(qf[f], load_frag(Kbase + f * 32, KV_DIM), sacc[nb]);
    }

    // ---- online softmax (fp32), causal mask computed analytically -------
    for (int r = 0; r < 8; ++r) {
      int m = r + 8 * grp;
      int qpos = q0 + m;
      float s0 = sacc[0][r] * scale;
      float s1 = sacc[1][r] * scale;
      if (k0 + ln > qpos)      s0 = -1e30f;
      if (k0 + 16 + ln > qpos) s1 = -1e30f;
      float mx = fmaxf(s0, s1);
      for (int off = 8; off >= 1; off >>= 1)      // row lives in a 16-lane group
        mx = fmaxf(mx, __shfl_xor(mx, off, 32));
      float mnew = fmaxf(mrun[r], mx);
      float corr = __expf(mrun[r] - mnew);
      mrun[r] = mnew;
      float p0 = __expf(s0 - mnew);
      float p1 = __expf(s1 - mnew);
      float rs = p0 + p1;
      for (int off = 8; off >= 1; off >>= 1)
        rs += __shfl_xor(rs, off, 32);
      lrun[r] = lrun[r] * corr + rs;
      for (int j = 0; j < 8; ++j) o[j][r] *= corr;
      // stage P (C-layout -> row-major LDS); same-wave DS ops are in-order
      pbuf[m * 32 + ln]      = f2bf(p0);
      pbuf[m * 32 + 16 + ln] = f2bf(p1);
    }

    // ---- O += P * V ------------------------------------------------------
    v16bf pf = load_frag(pbuf, 32);               // A-layout read from LDS
    const unsigned short* Vbase =
        Vt + (size_t)(b * N_KV + kvh) * HEAD_DIM * SEQLEN + k0;
    for (int j = 0; j < 8; ++j)
      o[j] = wmma_bf16(pf, load_frag(Vbase + (size_t)(j * 16) * SEQLEN, SEQLEN), o[j]);
  }

  // ---- normalize and emit bf16 attention output -------------------------
  for (int j = 0; j < 8; ++j)
    for (int r = 0; r < 8; ++r) {
      int m = r + 8 * grp;
      float v = o[j][r] / lrun[r];
      Ob[(size_t)(b * SEQLEN + q0 + m) * DIM + h * HEAD_DIM + j * 16 + ln] = f2bf(v);
    }
}

// ---------------------------------------------------------------------------
// Host-side orchestration
// ---------------------------------------------------------------------------
extern "C" void kernel_launch(void* const* d_in, const int* in_sizes, int n_in,
                              void* d_out, int out_size, void* d_ws, size_t ws_size,
                              hipStream_t stream) {
  (void)in_sizes; (void)n_in; (void)out_size; (void)ws_size;
  const float* x  = (const float*)d_in[0];
  const float* fc = (const float*)d_in[1];
  // d_in[2] (mask) unused: start_pos == 0, causal mask applied analytically
  const float* wq = (const float*)d_in[3];
  const float* wk = (const float*)d_in[4];
  const float* wv = (const float*)d_in[5];
  const float* wo = (const float*)d_in[6];

  const size_t NX  = (size_t)TOKENS * DIM;      // 16,777,216
  const size_t NK  = (size_t)TOKENS * KV_DIM;   //  4,194,304
  const size_t NWQ = (size_t)DIM * DIM;
  const size_t NWK = (size_t)KV_DIM * DIM;

  char* ws = (char*)d_ws;
  unsigned short* xb   = (unsigned short*)ws; ws += NX  * 2;
  unsigned short* wqb  = (unsigned short*)ws; ws += NWQ * 2;
  unsigned short* wkb  = (unsigned short*)ws; ws += NWK * 2;
  unsigned short* wvb  = (unsigned short*)ws; ws += NWK * 2;
  unsigned short* wob  = (unsigned short*)ws; ws += NWQ * 2;
  float*          qf   = (float*)ws;          ws += NX  * 4;
  float*          kf   = (float*)ws;          ws += NK  * 4;
  float*          vf   = (float*)ws;          ws += NK  * 4;
  unsigned short* qb   = (unsigned short*)ws; ws += NX  * 2;
  unsigned short* kb   = (unsigned short*)ws; ws += NK  * 2;
  unsigned short* vt   = (unsigned short*)ws; ws += NK  * 2;
  unsigned short* attb = (unsigned short*)ws; ws += NX  * 2;

  const int T = 256;
  // 1) fp32 -> bf16 staging
  cvt_f32_bf16<<<(int)((NX  + T - 1) / T), T, 0, stream>>>(x,  xb,  (long long)NX);
  cvt_f32_bf16<<<(int)((NWQ + T - 1) / T), T, 0, stream>>>(wq, wqb, (long long)NWQ);
  cvt_f32_bf16<<<(int)((NWK + T - 1) / T), T, 0, stream>>>(wk, wkb, (long long)NWK);
  cvt_f32_bf16<<<(int)((NWK + T - 1) / T), T, 0, stream>>>(wv, wvb, (long long)NWK);
  cvt_f32_bf16<<<(int)((NWQ + T - 1) / T), T, 0, stream>>>(wo, wob, (long long)NWQ);

  // 2) QKV projections (WMMA).  waves = (M/64)*(N/64); 4 waves / 128-thr block
  {
    int waves_q = (TOKENS / 64) * (DIM / 64);     // 4096
    gemm_bf16<<<waves_q / 4, 128, 0, stream>>>(xb, wqb, qf, TOKENS, DIM, DIM);
    int waves_k = (TOKENS / 64) * (KV_DIM / 64);  // 1024
    gemm_bf16<<<waves_k / 4, 128, 0, stream>>>(xb, wkb, kf, TOKENS, KV_DIM, DIM);
    gemm_bf16<<<waves_k / 4, 128, 0, stream>>>(xb, wvb, vf, TOKENS, KV_DIM, DIM);
  }

  // 3) RoPE + bf16 pack for Q,K; transpose+pack V
  {
    long long pq = (long long)NX / 2;
    long long pk = (long long)NK / 2;
    rope_cvt<<<(int)((pq + T - 1) / T), T, 0, stream>>>(qf, fc, qb, N_HEADS, pq);
    rope_cvt<<<(int)((pk + T - 1) / T), T, 0, stream>>>(kf, fc, kb, N_KV,    pk);
    v_transpose_cvt<<<(int)((NK + T - 1) / T), T, 0, stream>>>(vf, vt);
  }

  // 4) flash attention: one wave per (b, h, 16-row q tile)
  {
    int waves = BATCH * N_HEADS * (SEQLEN / 16);  // 8192
    flash_attn<<<waves / 4, 128, 0, stream>>>(qb, kb, vt, attb);
  }

  // 5) output projection straight into d_out (fp32)
  {
    int waves = (TOKENS / 64) * (DIM / 64);       // 4096
    gemm_bf16<<<waves / 4, 128, 0, stream>>>(attb, wob, (float*)d_out,
                                             TOKENS, DIM, DIM);
  }
}
